// Attention_12300786335934
// MI455X (gfx1250) — compile-verified, matrix operand order checked by reading, not run on hardware
//
#include <hip/hip_runtime.h>
#include <hip/hip_bf16.h>
#include <math.h>

typedef __attribute__((ext_vector_type(16))) __bf16 v16bf;
typedef __attribute__((ext_vector_type(8)))  __bf16 v8bf;
typedef __attribute__((ext_vector_type(8)))  float  v8f;

#define NB_WIN 54
#define NTOK   512
#define CDIM   512
#define NHEAD  8
#define DHEAD  64
#define NKEY   528    // 2 null + 512 tokens + 14 pad
#define NKEYP  544    // padded to multiple of 32 for WMMA K-steps
#define MTOT   27648  // NB_WIN * NTOK
#define NEGINF (-3.402823466e38f)

// ---------------- WMMA fragment helpers (ISA 7.12.2 wave32 layouts) ----------

// A/B fragment (16x32 bf16, row-major source with given element stride).
// lane<16 : row = lane, K in {8*hi..8*hi+7} U {16+8*hi..16+8*hi+7}, hi = lane>>4
static __device__ __forceinline__ v16bf frag_bf(const __bf16* base, int stride, int lane) {
  int row = lane & 15, hi = lane >> 4;
  const __bf16* p0 = base + (size_t)row * stride + 8 * hi;
  v8bf a = *(const v8bf*)(p0);
  v8bf b = *(const v8bf*)(p0 + 16);
  v16bf r;
#pragma unroll
  for (int i = 0; i < 8; i++) { r[i] = a[i]; r[i + 8] = b[i]; }
  return r;
}

// Same fragment, sourced from f32 (LDS softmax probabilities), converted to bf16.
static __device__ __forceinline__ v16bf frag_f32(const float* base, int stride, int lane) {
  int row = lane & 15, hi = lane >> 4;
  const float* p0 = base + (size_t)row * stride + 8 * hi;
  v16bf r;
#pragma unroll
  for (int i = 0; i < 8; i++) { r[i] = (__bf16)p0[i]; r[i + 8] = (__bf16)p0[16 + i]; }
  return r;
}

static __device__ __forceinline__ v8f wmma_bf16(v16bf a, v16bf b, v8f c) {
  return __builtin_amdgcn_wmma_f32_16x16x32_bf16(false, a, false, b, (short)0, c, false, false);
}

static __device__ __forceinline__ float lrelu(float v) { return v > 0.f ? v : 0.1f * v; }

// window-row -> flat (b, n) base index into (B, 13824) token space
static __device__ __forceinline__ size_t token_global_base(int row) {
  int win = row >> 9, t = row & 511;
  int b = win / 27, wr = win % 27;
  int wx = wr / 9, wy = (wr / 3) % 3, wz = wr % 3;
  int lx = t >> 6, ly = (t >> 3) & 7, lz = t & 7;
  int gx = wx * 8 + lx, gy = wy * 8 + ly, gz = wz * 8 + lz;
  int n = (gx * 24 + gy) * 24 + gz;
  return (size_t)b * 13824 + (size_t)n;
}

// ---------------- K0: convert + transpose weights to bf16 --------------------
__global__ void k_convert(const float* __restrict__ Wq, const float* __restrict__ Wkv,
                          const float* __restrict__ Wout, const float* __restrict__ W2,
                          __bf16* wtqkv, __bf16* woutT, __bf16* w2t) {
  int i = blockIdx.x * 256 + threadIdx.x;
  if (i < 1536 * 512) {                       // [Wq|Wkv]^T : [1536][512]
    int n = i >> 9, kk = i & 511;
    float v = (n < 512) ? Wq[(size_t)kk * 512 + n] : Wkv[(size_t)kk * 1024 + (n - 512)];
    wtqkv[(size_t)n * 512 + kk] = (__bf16)v;
  } else if (i < 1536 * 512 + 512 * 512) {    // Wout^T : [512][512]
    int j = i - 1536 * 512; int n = j >> 9, kk = j & 511;
    woutT[(size_t)n * 512 + kk] = (__bf16)Wout[(size_t)kk * 512 + n];
  } else {                                    // pb_w2^T : [256][256]
    int j = i - 1536 * 512 - 512 * 512; int n = j >> 8, kk = j & 255;
    w2t[(size_t)n * 256 + kk] = (__bf16)W2[(size_t)kk * 256 + n];
  }
}

// ---------------- K1: LayerNorm + window partition ---------------------------
__global__ void k_ln(const float* __restrict__ x, const unsigned char* __restrict__ mask,
                     const float* __restrict__ gamma, __bf16* xn, unsigned char* tmw) {
  int row = blockIdx.x;                    // win*512 + t
  size_t gbase = token_global_base(row);
  const float* xr = x + gbase * 512;
  int tid = threadIdx.x;
  float v0 = xr[tid], v1 = xr[tid + 256];
  __shared__ float red[256];
  red[tid] = v0 + v1; __syncthreads();
  for (int s = 128; s > 0; s >>= 1) { if (tid < s) red[tid] += red[tid + s]; __syncthreads(); }
  float mu = red[0] * (1.0f / 512.0f);
  __syncthreads();
  float d0 = v0 - mu, d1 = v1 - mu;
  red[tid] = d0 * d0 + d1 * d1; __syncthreads();
  for (int s = 128; s > 0; s >>= 1) { if (tid < s) red[tid] += red[tid + s]; __syncthreads(); }
  float rstd = rsqrtf(red[0] * (1.0f / 512.0f) + 1e-5f);
  xn[(size_t)row * 512 + tid]       = (__bf16)(d0 * rstd * gamma[tid]);
  xn[(size_t)row * 512 + tid + 256] = (__bf16)(d1 * rstd * gamma[tid + 256]);
  if (tid == 0) tmw[row] = mask[gbase];
}

// ---------------- K2: positional-bias MLP (WMMA middle layer) ----------------
// Per-wave: ONE m-tile (8 preloaded A-fragments kept live, ~64 VGPRs) and four
// sequential n-tiles with a single 8-VGPR accumulator -> no scratch spills.
__global__ void k_posbias(const float* __restrict__ W1, const float* __restrict__ b1,
                          const __bf16* __restrict__ w2t, const float* __restrict__ b2,
                          const float* __restrict__ W3p, const float* __restrict__ b3,
                          float* bias) {
  __shared__ __bf16 h1[32 * 256];
  __shared__ __bf16 h2[32 * 256];
  int tid = threadIdx.x, lane = tid & 31, wave = tid >> 5;
  int pg0 = blockIdx.x * 32;

  // layer 1: K=3, scalar. thread -> pair p = tid>>3, outputs ob..ob+31
  {
    int p = tid >> 3; int ob = (tid & 7) * 32;
    int pg = pg0 + p; int qi = pg >> 9, kj = pg & 511;
    float r[3];
    int ci[3] = { qi >> 6, (qi >> 3) & 7, qi & 7 };
    int cj[3] = { kj >> 6, (kj >> 3) & 7, kj & 7 };
#pragma unroll
    for (int d = 0; d < 3; d++) {
      float rel = (float)(ci[d] - cj[d]);
      float sg = rel > 0.f ? 1.f : (rel < 0.f ? -1.f : 0.f);
      r[d] = sg * log1pf(fabsf(rel));
    }
    for (int o = 0; o < 32; o++) {
      int oo = ob + o;
      float hv = r[0] * W1[oo] + r[1] * W1[256 + oo] + r[2] * W1[512 + oo] + b1[oo];
      h1[p * 256 + oo] = (__bf16)lrelu(hv);
    }
  }
  __syncthreads();

  // layer 2: 32x256 @ 256x256 via WMMA. wave -> mt = wave&1, nt = (wave>>1)+4g
  {
    int mt = wave & 1, ntb = wave >> 1;
    v16bf afr[8];
#pragma unroll
    for (int kc = 0; kc < 8; kc++)
      afr[kc] = frag_bf(h1 + mt * 16 * 256 + kc * 32, 256, lane);
#pragma unroll
    for (int g = 0; g < 4; g++) {
      int nt = ntb + g * 4;
      v8f c = {};
#pragma unroll
      for (int kc = 0; kc < 8; kc++) {
        v16bf b = frag_bf(w2t + (size_t)(nt * 16) * 256 + kc * 32, 256, lane);
        c = wmma_bf16(afr[kc], b, c);
      }
      int hi = lane >> 4, n15 = lane & 15;
      float bb = b2[nt * 16 + n15];
#pragma unroll
      for (int rr = 0; rr < 8; rr++) {
        int m = mt * 16 + rr + 8 * hi;
        h2[m * 256 + nt * 16 + n15] = (__bf16)lrelu(c[rr] + bb);
      }
    }
  }
  __syncthreads();

  // layer 3: N=8, scalar
  {
    int p = tid >> 3, hh = tid & 7;
    float s = b3[hh];
    for (int kk = 0; kk < 256; kk++) s += (float)h2[p * 256 + kk] * W3p[kk * 8 + hh];
    int pg = pg0 + p;
    bias[(size_t)hh * 262144 + pg] = s;   // bias[h][qi][kj], pg = qi*512+kj
  }
}

// ---------------- K3: QKV projection GEMM (27648x512 @ 512x1536) -------------
__global__ void k_gemm_qkv(const __bf16* __restrict__ A, const __bf16* __restrict__ BT,
                           float* __restrict__ Cout) {
  int bm = blockIdx.x, bn = blockIdx.y;
  int tid = threadIdx.x, lane = tid & 31, wave = tid >> 5;
  int mw = wave & 3, nw = wave >> 2;
  const __bf16* Ab = A  + ((size_t)bm * 128 + mw * 32) * 512;
  const __bf16* Bb = BT + ((size_t)bn * 128 + nw * 64) * 512;
  v8f c[2][4] = {};
  for (int kc = 0; kc < 16; kc++) {
    v16bf a0 = frag_bf(Ab + kc * 32, 512, lane);
    v16bf a1 = frag_bf(Ab + (size_t)16 * 512 + kc * 32, 512, lane);
    v16bf b0 = frag_bf(Bb + kc * 32, 512, lane);
    v16bf b1 = frag_bf(Bb + (size_t)16 * 512 + kc * 32, 512, lane);
    v16bf b2 = frag_bf(Bb + (size_t)32 * 512 + kc * 32, 512, lane);
    v16bf b3 = frag_bf(Bb + (size_t)48 * 512 + kc * 32, 512, lane);
    c[0][0] = wmma_bf16(a0, b0, c[0][0]); c[0][1] = wmma_bf16(a0, b1, c[0][1]);
    c[0][2] = wmma_bf16(a0, b2, c[0][2]); c[0][3] = wmma_bf16(a0, b3, c[0][3]);
    c[1][0] = wmma_bf16(a1, b0, c[1][0]); c[1][1] = wmma_bf16(a1, b1, c[1][1]);
    c[1][2] = wmma_bf16(a1, b2, c[1][2]); c[1][3] = wmma_bf16(a1, b3, c[1][3]);
  }
  int hi = lane >> 4, n15 = lane & 15;
#pragma unroll
  for (int i = 0; i < 2; i++)
#pragma unroll
    for (int j = 0; j < 4; j++)
#pragma unroll
      for (int rr = 0; rr < 8; rr++) {
        int row = bm * 128 + mw * 32 + i * 16 + rr + 8 * hi;
        int col = bn * 128 + nw * 64 + j * 16 + n15;
        Cout[(size_t)row * 1536 + col] = c[i][j][rr];
      }
}

// ---------------- K4: l2norm + scales, layout q/k/v^T ------------------------
__global__ void k_post(const float* __restrict__ qkv, const float* __restrict__ qs,
                       const float* __restrict__ ks, __bf16* q, __bf16* k, __bf16* vt) {
  int row = blockIdx.x;                 // win*512 + t
  int win = row >> 9, t = row & 511;
  int tid = threadIdx.x;                // 512 threads
  int h = tid >> 6, d = tid & 63;
  const float* base = qkv + (size_t)row * 1536;
  float qv = base[h * 64 + d];
  float kv = base[512 + h * 64 + d];
  float vv = base[1024 + h * 64 + d];
  __shared__ float red[512];
  __shared__ float nrm[8];
  red[tid] = qv * qv; __syncthreads();
  if (d == 0) { float s = 0; for (int i = 0; i < 64; i++) s += red[h * 64 + i];
                nrm[h] = fmaxf(sqrtf(s), 1e-12f); }
  __syncthreads();
  float qn = qv / nrm[h] * qs[d] * 8.0f;     // SCALE folded into q
  red[tid] = kv * kv; __syncthreads();
  if (d == 0) { float s = 0; for (int i = 0; i < 64; i++) s += red[h * 64 + i];
                nrm[h] = fmaxf(sqrtf(s), 1e-12f); }
  __syncthreads();
  float kn = kv / nrm[h] * ks[d];
  size_t wh = (size_t)win * 8 + h;
  q[(wh * 512 + t) * 64 + d]          = (__bf16)qn;
  k[(wh * NKEY + 2 + t) * 64 + d]     = (__bf16)kn;
  vt[(wh * 64 + d) * NKEYP + 2 + t]   = (__bf16)vv;
}

// ---------------- K5: null-KV rows + zero padding ----------------------------
__global__ void k_fill(const float* __restrict__ null_kv, const float* __restrict__ ks,
                       __bf16* k, __bf16* vt) {
  int wh = blockIdx.x;                  // win*8 + h, 0..431
  int h = wh & 7;
  int tid = threadIdx.x;
  if (tid < 64) {
    int d = tid;
#pragma unroll
    for (int nn = 0; nn < 2; nn++) {
      const float* nk = null_kv + h * 256 + nn * 128;   // (H, NNK, 2, DH) k-part
      const float* nv = nk + 64;                         // v-part
      float s = 0; for (int i = 0; i < 64; i++) s += nk[i] * nk[i];
      float inv = 1.0f / fmaxf(sqrtf(s), 1e-12f);
      k[((size_t)wh * NKEY + nn) * 64 + d]   = (__bf16)(nk[d] * inv * ks[d]);
      vt[((size_t)wh * 64 + d) * NKEYP + nn] = (__bf16)(nv[d]);
    }
  }
  for (int i = tid; i < 14 * 64; i += 256) {             // k rows 514..527 -> 0
    int rr = 514 + i / 64, d = i & 63;
    k[((size_t)wh * NKEY + rr) * 64 + d] = (__bf16)0.f;
  }
  for (int i = tid; i < 64 * 30; i += 256) {             // vt cols 514..543 -> 0
    int d = i / 30, cc = 514 + i % 30;
    vt[((size_t)wh * 64 + d) * NKEYP + cc] = (__bf16)0.f;
  }
}

// ---------------- K6: fused attention per (window, head, 64-query tile) ------
__global__ void k_attn(const __bf16* __restrict__ q, const __bf16* __restrict__ k,
                       const __bf16* __restrict__ vt, const float* __restrict__ bias,
                       const unsigned char* __restrict__ tmw, __bf16* attn_out) {
  extern __shared__ float sim[];        // 64 x 544 f32 (139 KB, CDNA5 LDS)
  __shared__ float red4[64 * 4];        // softmax cross-thread reduction
  int blk = blockIdx.x;
  int win = blk / 64; int rem = blk % 64; int h = rem >> 3; int qb = rem & 7;
  size_t wh = (size_t)win * 8 + h;
  int tid = threadIdx.x, lane = tid & 31, wave = tid >> 5;

  for (int i = tid; i < 64 * 16; i += 256)               // zero K-pad cols 528..543
    sim[(i >> 4) * NKEYP + 528 + (i & 15)] = 0.f;

  const __bf16* qbase = q + (wh * 512 + (size_t)qb * 64) * 64;
  const __bf16* kbase = k + wh * NKEY * 64;

  // sim = q @ k^T (+ bias, key mask) -> LDS
  for (int nt = wave; nt < 33; nt += 8) {
    int j = nt * 16 + (lane & 15);
    bool kmask; float bscale = 0.f;
    if (j < 2) { kmask = true; }
    else if (j < 514) { kmask = (tmw[win * 512 + (j - 2)] != 0); bscale = 1.f; }
    else kmask = false;
    for (int mt = 0; mt < 4; mt++) {
      v8f c = {};
#pragma unroll
      for (int kc = 0; kc < 2; kc++) {
        v16bf a = frag_bf(qbase + mt * 16 * 64 + kc * 32, 64, lane);
        v16bf b = frag_bf(kbase + (size_t)(nt * 16) * 64 + kc * 32, 64, lane);
        c = wmma_bf16(a, b, c);
      }
      int hi = lane >> 4;
#pragma unroll
      for (int rr = 0; rr < 8; rr++) {
        int m = mt * 16 + rr + 8 * hi;
        int qi = qb * 64 + m;
        float v;
        if (kmask) {
          float bb = (bscale != 0.f) ? bias[((size_t)h * 512 + qi) * 512 + (j - 2)] : 0.f;
          v = c[rr] + bb;
        } else v = NEGINF;
        sim[m * NKEYP + j] = v;
      }
    }
  }
  __syncthreads();

  // row softmax over 528 keys: 4 threads per row, 132 columns each
  {
    int r = tid >> 2, qp = tid & 3;
    float* row = sim + r * NKEYP;
    int j0 = qp * 132, j1 = j0 + 132;
    float m = NEGINF;
    for (int j = j0; j < j1; j++) m = fmaxf(m, row[j]);
    red4[r * 4 + qp] = m; __syncthreads();
    float mx = fmaxf(fmaxf(red4[r * 4], red4[r * 4 + 1]),
                     fmaxf(red4[r * 4 + 2], red4[r * 4 + 3]));
    float s = 0.f;
    for (int j = j0; j < j1; j++) { float e = __expf(row[j] - mx); row[j] = e; s += e; }
    __syncthreads();
    red4[r * 4 + qp] = s; __syncthreads();
    float inv = 1.0f / (red4[r * 4] + red4[r * 4 + 1] + red4[r * 4 + 2] + red4[r * 4 + 3]);
    for (int j = j0; j < j1; j++) row[j] *= inv;
  }
  __syncthreads();

  // out = P @ V  (P: LDS f32 -> bf16 fragments; V read transposed)
  for (int ti = wave; ti < 16; ti += 8) {
    int mt = ti & 3, dt = ti >> 2;
    v8f c = {};
    for (int kc = 0; kc < 17; kc++) {
      v16bf a = frag_f32(sim + mt * 16 * NKEYP + kc * 32, NKEYP, lane);
      v16bf b = frag_bf(vt + (wh * 64 + (size_t)dt * 16) * NKEYP + kc * 32, NKEYP, lane);
      c = wmma_bf16(a, b, c);
    }
    int hi = lane >> 4, n15 = lane & 15;
#pragma unroll
    for (int rr = 0; rr < 8; rr++) {
      int m = mt * 16 + rr + 8 * hi;
      int qi = qb * 64 + m;
      float qm = (tmw[win * 512 + qi] != 0) ? 1.f : 0.f;
      attn_out[((size_t)win * 512 + qi) * 512 + h * 64 + dt * 16 + n15] = (__bf16)(c[rr] * qm);
    }
  }
}

// ---------------- K7: output projection + reverse partition ------------------
__global__ void k_gemm_out(const __bf16* __restrict__ A, const __bf16* __restrict__ BT,
                           float* __restrict__ out) {
  int bm = blockIdx.x, bn = blockIdx.y;
  int tid = threadIdx.x, lane = tid & 31, wave = tid >> 5;
  int mw = wave & 3, nw = wave >> 2;
  const __bf16* Ab = A  + ((size_t)bm * 128 + mw * 32) * 512;
  const __bf16* Bb = BT + ((size_t)bn * 128 + nw * 64) * 512;
  v8f c[2][4] = {};
  for (int kc = 0; kc < 16; kc++) {
    v16bf a0 = frag_bf(Ab + kc * 32, 512, lane);
    v16bf a1 = frag_bf(Ab + (size_t)16 * 512 + kc * 32, 512, lane);
    v16bf b0 = frag_bf(Bb + kc * 32, 512, lane);
    v16bf b1 = frag_bf(Bb + (size_t)16 * 512 + kc * 32, 512, lane);
    v16bf b2 = frag_bf(Bb + (size_t)32 * 512 + kc * 32, 512, lane);
    v16bf b3 = frag_bf(Bb + (size_t)48 * 512 + kc * 32, 512, lane);
    c[0][0] = wmma_bf16(a0, b0, c[0][0]); c[0][1] = wmma_bf16(a0, b1, c[0][1]);
    c[0][2] = wmma_bf16(a0, b2, c[0][2]); c[0][3] = wmma_bf16(a0, b3, c[0][3]);
    c[1][0] = wmma_bf16(a1, b0, c[1][0]); c[1][1] = wmma_bf16(a1, b1, c[1][1]);
    c[1][2] = wmma_bf16(a1, b2, c[1][2]); c[1][3] = wmma_bf16(a1, b3, c[1][3]);
  }
  int hi = lane >> 4, n15 = lane & 15;
#pragma unroll
  for (int i = 0; i < 2; i++)
#pragma unroll
    for (int j = 0; j < 4; j++)
#pragma unroll
      for (int rr = 0; rr < 8; rr++) {
        int row = bm * 128 + mw * 32 + i * 16 + rr + 8 * hi;
        int col = bn * 128 + nw * 64 + j * 16 + n15;
        out[token_global_base(row) * 512 + col] = c[i][j][rr];
      }
}

// ---------------- launcher ---------------------------------------------------
extern "C" void kernel_launch(void* const* d_in, const int* in_sizes, int n_in,
                              void* d_out, int out_size, void* d_ws, size_t ws_size,
                              hipStream_t stream) {
  (void)in_sizes; (void)n_in; (void)out_size; (void)ws_size;
  const float*         x      = (const float*)d_in[0];
  const unsigned char* mask   = (const unsigned char*)d_in[1];
  const float*         gamma  = (const float*)d_in[2];
  const float*         Wq     = (const float*)d_in[3];
  const float*         Wkv    = (const float*)d_in[4];
  const float*         Wout   = (const float*)d_in[5];
  const float*         qsc    = (const float*)d_in[6];
  const float*         ksc    = (const float*)d_in[7];
  const float*         nullkv = (const float*)d_in[8];
  const float*         pw1    = (const float*)d_in[9];
  const float*         pb1    = (const float*)d_in[10];
  const float*         pw2    = (const float*)d_in[11];
  const float*         pb2    = (const float*)d_in[12];
  const float*         pw3    = (const float*)d_in[13];
  const float*         pb3    = (const float*)d_in[14];

  char* ws = (char*)d_ws;
  size_t off = 0;
  auto alloc = [&](size_t bytes) { size_t r = off; off = (off + bytes + 255) & ~(size_t)255; return r; };
  __bf16* xn      = (__bf16*)(ws + alloc((size_t)MTOT * 512 * 2));
  unsigned char* tmw = (unsigned char*)(ws + alloc(MTOT));
  __bf16* wtqkv   = (__bf16*)(ws + alloc((size_t)1536 * 512 * 2));
  __bf16* woutT   = (__bf16*)(ws + alloc((size_t)512 * 512 * 2));
  __bf16* w2t     = (__bf16*)(ws + alloc((size_t)256 * 256 * 2));
  float*  qkv     = (float*)(ws + alloc((size_t)MTOT * 1536 * 4));
  __bf16* qb      = (__bf16*)(ws + alloc((size_t)NB_WIN * 8 * 512 * 64 * 2));
  __bf16* kb      = (__bf16*)(ws + alloc((size_t)NB_WIN * 8 * NKEY * 64 * 2));
  __bf16* vtb     = (__bf16*)(ws + alloc((size_t)NB_WIN * 8 * 64 * NKEYP * 2));
  float*  bias    = (float*)(ws + alloc((size_t)8 * 512 * 512 * 4));
  __bf16* attn    = (__bf16*)(ws + alloc((size_t)MTOT * 512 * 2));
  float*  out     = (float*)d_out;

  k_convert<<<4352, 256, 0, stream>>>(Wq, Wkv, Wout, pw2, wtqkv, woutT, w2t);
  k_ln<<<MTOT, 256, 0, stream>>>(x, mask, gamma, xn, tmw);
  k_posbias<<<8192, 256, 0, stream>>>(pw1, pb1, w2t, pb2, pw3, pb3, bias);
  k_gemm_qkv<<<dim3(216, 12), 256, 0, stream>>>(xn, wtqkv, qkv);
  k_post<<<MTOT, 512, 0, stream>>>(qkv, qsc, ksc, qb, kb, vtb);
  k_fill<<<432, 256, 0, stream>>>(nullkv, ksc, kb, vtb);
  k_attn<<<NB_WIN * 64, 256, 64 * NKEYP * 4, stream>>>(qb, kb, vtb, bias, tmw, attn);
  k_gemm_out<<<dim3(216, 4), 256, 0, stream>>>(attn, woutT, out);
}
